// LSTM_85160611545138
// MI455X (gfx1250) — matrix operand order repaired and checked by reading, not compiled, not run on hardware
//
#include <hip/hip_runtime.h>

// ---------------------------------------------------------------------------
// LSTM on MI455X (gfx1250): batch-partitioned fused scan with bf16 WMMA.
//   B=256, T=1024, I=128, H=256, O=128
// 16 workgroups (16 batch rows each) run the full T=1024 recurrence privately:
//   - cell state c lives in f32 WMMA accumulator registers (per wave, 32 cols)
//   - hidden state h exchanged via double-buffered LDS bf16 tile per step
//   - bf16 weights stream from L2 every step (0.85 MB, shared by all WGs),
//     software-pipelined (tile p+1 loads vs tile p WMMAs) and shaped with
//     sched_group_barrier so ~24 loads stay in flight behind each WMMA block.
// Per-lane 64-bit base addresses are made opaque with an empty asm each
// iteration: blocks LICM (no hoist+spill). Every fragment load is a
// global_load_b128 off the per-lane base with a compile-time immediate
// offset (no per-iteration address VALU). AS(1) casts keep loads GLOBAL.
// ---------------------------------------------------------------------------

typedef __attribute__((ext_vector_type(16))) __bf16 v16bf;
typedef __attribute__((ext_vector_type(8)))  __bf16 v8bf;
typedef __attribute__((ext_vector_type(8)))  float  v8f;
typedef __attribute__((ext_vector_type(4)))  float  v4f;

#define SEQ_LEN 1024
#define IN_DIM  128
#define HID     256
#define OUT_DIM 128
#define BATCH   256
#define HPAD    264   // 16B-aligned row pad -> LDS banks stride by 4 per row

#if __has_builtin(__builtin_amdgcn_sched_group_barrier)
// mask: 0x0008 = MFMA/WMMA, 0x0020 = VMEM read, 0x0100 = DS read
#define SCHED_GROUP(mask, size, id) \
  __builtin_amdgcn_sched_group_barrier((mask), (size), (id))
#else
#define SCHED_GROUP(mask, size, id)
#endif

// ----------------------------- helpers -------------------------------------

__global__ void f32_to_bf16_kernel(const float* __restrict__ src,
                                   __bf16* __restrict__ dst, int n) {
  int i = blockIdx.x * blockDim.x + threadIdx.x;
  if (i < n) dst[i] = (__bf16)src[i];
}

__device__ __forceinline__ float fast_sigmoid(float x) {
  return 1.0f / (1.0f + __expf(-x));
}
__device__ __forceinline__ float fast_tanh(float x) {
  float e = __expf(-2.0f * x);
  return (1.0f - e) / (1.0f + e);
}

__device__ __forceinline__ v8f wmma_bf16(v16bf a, v16bf b, v8f c) {
  // D = A(16x32 bf16) * B(32x16 bf16) + C(16x16 f32)
  return __builtin_amdgcn_wmma_f32_16x16x32_bf16(
      /*neg_a=*/false, a, /*neg_b=*/false, b,
      /*c_mod=*/(short)0, c, /*reuse_a=*/false, /*reuse_b=*/false);
}

union AFrag { v8bf h[2]; v16bf v; };
union F32x8 { v4f q[2]; v8f v; };

// A-fragment (bf16 16x32): per-lane interleaved 8-chunks: [kb..kb+7] and
// [kb+16..kb+23]; caller folds the lane-half (+8) into kb.
__device__ __forceinline__ v16bf load_afrag(const __bf16* row, int kb) {
  AFrag a;
  a.h[0] = *(const v8bf*)(row + kb);
  a.h[1] = *(const v8bf*)(row + kb + 16);
  return a.v;
}

// Global (AS1) loads from a per-lane 64-bit address; the constant part of
// `addr` folds into each instruction's signed 24-bit immediate offset.
// Multi-dword payloads are issued as independent b128 loads so EVERY half
// keeps immediate-offset addressing off the same base register (no v_add_u64).
__device__ __forceinline__ v16bf gload_bfrag(unsigned long long addr) {
  typedef __attribute__((address_space(1))) const v8bf* gvp;
  AFrag a;
  a.h[0] = *(gvp)addr;
  a.h[1] = *(gvp)(addr + 32);
  return a.v;
}
__device__ __forceinline__ v8f gload_f32x8(unsigned long long addr) {
  typedef __attribute__((address_space(1))) const v4f* gvp;
  F32x8 f;
  f.q[0] = *(gvp)addr;
  f.q[1] = *(gvp)(addr + 16);
  return f.v;
}

// ----------------------------- scan kernel ---------------------------------

__launch_bounds__(256, 1)
__global__ void lstm_scan_kernel(
    const float*  __restrict__ x,    // [B][T][I] f32
    const __bf16* __restrict__ wx,   // [4][HID][IN_DIM] bf16 (g,i,f,o)
    const __bf16* __restrict__ wh,   // [4][HID][HID]    bf16
    const __bf16* __restrict__ wp,   // [OUT_DIM][HID]   bf16
    const float*  __restrict__ bgx, const float* __restrict__ bix,
    const float*  __restrict__ bfx, const float* __restrict__ box,
    const float*  __restrict__ bp,
    float*        __restrict__ out)  // [B][OUT_DIM] f32
{
  __shared__ __bf16 hbuf[2][16][HPAD];   // double-buffered h tile (bf16)

  const int tid   = threadIdx.x;
  const int wave  = tid >> 5;            // 8 waves
  const int lane  = tid & 31;
  const int l16   = lane & 15;
  const int lhalf = lane >> 4;           // 0: lanes 0-15, 1: lanes 16-31
  const int m0    = blockIdx.x * 16;     // this WG's batch rows
  const int n0    = wave * 32;           // this wave's hidden columns

  // zero the t=0 hidden buffer
  for (int i = tid; i < 16 * HPAD; i += 256)
    (&hbuf[0][0][0])[i] = (__bf16)0.0f;
  __syncthreads();

  // per-lane gate biases (depend on column only -> broadcast over rows)
  const float* bptr[4] = {bgx, bix, bfx, box};
  float bias[4][2];
#pragma unroll
  for (int g = 0; g < 4; ++g)
#pragma unroll
    for (int j = 0; j < 2; ++j)
      bias[g][j] = bptr[g][n0 + j * 16 + l16];

  v8f cst[2] = {v8f{}, v8f{}};           // cell state slice, f32, in VGPRs
  int cur = 0;
  const int klo = lhalf * 8;             // A-frag lane-half K offset

  // Per-lane base addresses: divergent parts (lane column / lane half / row)
  // folded in ONCE; everything else is a compile-time immediate per load.
  unsigned long long wxl = (unsigned long long)(uintptr_t)wx +
      (unsigned)(((n0 + l16) * IN_DIM + lhalf * 8) * 2);
  unsigned long long whl = (unsigned long long)(uintptr_t)wh +
      (unsigned)(((n0 + l16) * HID + lhalf * 8) * 2);
  unsigned long long xl = (unsigned long long)(uintptr_t)x +
      ((unsigned long long)(m0 + l16) * SEQ_LEN * IN_DIM + (unsigned)klo) * 4ull;

  for (int t = 0; t < SEQ_LEN; ++t) {
    // Opaque rewrite: compiler cannot prove the bases loop-invariant, so the
    // 96 B-fragment loads stay in-loop (streaming from L2) instead of being
    // hoisted into ~768 VGPRs + scratch spills.
    asm volatile("" : "+v"(wxl), "+v"(whl), "+v"(xl));

    // B-fragment loader for output tile p (p = g*2 + j): 4 Wx + 8 Wh chunks.
    auto load_tile = [&](int p, v16bf* dst) {
      const int g = p >> 1, j = p & 1;
      const unsigned wxo = (unsigned)(((g * HID + j * 16) * IN_DIM) * 2);
#pragma unroll
      for (int kc = 0; kc < IN_DIM / 32; ++kc)
        dst[kc] = gload_bfrag(wxl + (wxo + kc * 64));
      const unsigned who = (unsigned)(((g * HID + j * 16) * HID) * 2);
#pragma unroll
      for (int kc = 0; kc < HID / 32; ++kc)
        dst[4 + kc] = gload_bfrag(whl + (who + kc * 64));
    };

    // ---- start streaming tile 0's weights immediately
    v16bf bfr[2][12];                      // double-buffered B fragments
    load_tile(0, bfr[0]);

    // ---- x_t A-fragments: K=128 -> 4 chunks; load f32, convert to bf16
    if (t + 1 < SEQ_LEN)
      __builtin_prefetch((const void*)(uintptr_t)(xl + IN_DIM * 4), 0, 1);
    v16bf xa[IN_DIM / 32];
#pragma unroll
    for (int kc = 0; kc < IN_DIM / 32; ++kc) {
      v8f f0 = gload_f32x8(xl + (unsigned)(kc * 128));
      v8f f1 = gload_f32x8(xl + (unsigned)(kc * 128 + 64));
      AFrag a;
#pragma unroll
      for (int i = 0; i < 8; ++i) {
        a.h[0][i] = (__bf16)f0[i];
        a.h[1][i] = (__bf16)f1[i];
      }
      xa[kc] = a.v;
    }

    // ---- h_{t-1} A-fragments from LDS: K=256 -> 8 chunks
    const __bf16* hrow = hbuf[cur][l16];
    v16bf ha[HID / 32];
#pragma unroll
    for (int kc = 0; kc < HID / 32; ++kc)
      ha[kc] = load_afrag(hrow, kc * 32 + klo);

    // Prologue pipeline spec: tile0 fragments (24 vmem) + x tile (16 vmem),
    // then the 16 LDS A-fragment reads.
    SCHED_GROUP(0x0020, 40, 0);
    SCHED_GROUP(0x0100, 16, 0);

    // ---- gate pre-activations: 8 output tiles (4 gates x 2 N-tiles),
    //      12 WMMAs each, software-pipelined against tile p+1's loads
    v8f z[4][2];
#pragma unroll
    for (int p = 0; p < 8; ++p) {
      if (p + 1 < 8) load_tile(p + 1, bfr[(p + 1) & 1]);
      const v16bf* bf = bfr[p & 1];
      v8f acc = {};                        // first WMMA gets inline C=0
#pragma unroll
      for (int kc = 0; kc < IN_DIM / 32; ++kc)
        acc = wmma_bf16(xa[kc], bf[kc], acc);
#pragma unroll
      for (int kc = 0; kc < HID / 32; ++kc)
        acc = wmma_bf16(ha[kc], bf[4 + kc], acc);
      z[p >> 1][p & 1] = acc;
      // Pipeline shape: keep tile p+1's 24 fragment loads in flight ahead of
      // tile p's 12 WMMAs.
      if (p + 1 < 8) SCHED_GROUP(0x0020, 24, 0);
      SCHED_GROUP(0x0008, 12, 0);
    }

    // ---- elementwise LSTM cell update (bias folded in here);
    //      write new h (bf16) to the other LDS buffer
    const int nxt = cur ^ 1;
#pragma unroll
    for (int j = 0; j < 2; ++j) {
      v8f hnew;
#pragma unroll
      for (int i = 0; i < 8; ++i) {
        float gg = fast_tanh(z[0][j][i] + bias[0][j]);
        float ig = fast_sigmoid(z[1][j][i] + bias[1][j]);
        float fg = fast_sigmoid(z[2][j][i] + bias[2][j]);
        float og = fast_sigmoid(z[3][j][i] + bias[3][j]);
        float c  = gg * ig + cst[j][i] * fg;
        cst[j][i] = c;
        hnew[i]  = fast_tanh(c) * og;
      }
      const int col = n0 + j * 16 + l16;
#pragma unroll
      for (int r = 0; r < 8; ++r)           // C layout: VGPR r -> row r+8*lhalf
        hbuf[nxt][r + 8 * lhalf][col] = (__bf16)hnew[r];
    }
    __syncthreads();
    cur = nxt;

    xl += IN_DIM * 4;                       // advance x row by one timestep
  }

  // ---- final projection: out = h_T @ Wp^T + bp ; wave w -> out cols [16w,16w+16)
  {
    const __bf16* hrow = hbuf[cur][l16];
    const int ocol = wave * 16 + l16;
    v8f acc = {};
    unsigned long long wpl = (unsigned long long)(uintptr_t)wp +
        (unsigned)((ocol * HID + lhalf * 8) * 2);
#pragma unroll
    for (int kc = 0; kc < HID / 32; ++kc)
      acc = wmma_bf16(load_afrag(hrow, kc * 32 + klo),
                      gload_bfrag(wpl + (unsigned)(kc * 64)), acc);
    const float b = bp[ocol];
#pragma unroll
    for (int r = 0; r < 8; ++r)
      out[(size_t)(m0 + r + 8 * lhalf) * OUT_DIM + wave * 16 + l16] = acc[r] + b;
  }
}

// ----------------------------- launcher ------------------------------------

extern "C" void kernel_launch(void* const* d_in, const int* in_sizes, int n_in,
                              void* d_out, int out_size, void* d_ws, size_t ws_size,
                              hipStream_t stream) {
  (void)in_sizes; (void)n_in; (void)out_size; (void)ws_size;

  const float* x = (const float*)d_in[0];

  // workspace layout: bf16 weights (total ~0.85 MB)
  __bf16* wx = (__bf16*)d_ws;                 // [4][HID][IN_DIM]
  __bf16* wh = wx + 4 * HID * IN_DIM;         // [4][HID][HID]
  __bf16* wp = wh + 4 * HID * HID;            // [OUT_DIM][HID]

  auto cvt = [&](const void* src, __bf16* dst, int n) {
    f32_to_bf16_kernel<<<(n + 255) / 256, 256, 0, stream>>>(
        (const float*)src, dst, n);
  };
  // gate order g,i,f,o : d_in = {x, Wgx,bgx,Wgh, Wix,bix,Wih, Wfx,bfx,Wfh,
  //                              Wox,box,Woh, Wp, bp}
  cvt(d_in[1],  wx + 0 * HID * IN_DIM, HID * IN_DIM);
  cvt(d_in[4],  wx + 1 * HID * IN_DIM, HID * IN_DIM);
  cvt(d_in[7],  wx + 2 * HID * IN_DIM, HID * IN_DIM);
  cvt(d_in[10], wx + 3 * HID * IN_DIM, HID * IN_DIM);
  cvt(d_in[3],  wh + 0 * HID * HID, HID * HID);
  cvt(d_in[6],  wh + 1 * HID * HID, HID * HID);
  cvt(d_in[9],  wh + 2 * HID * HID, HID * HID);
  cvt(d_in[12], wh + 3 * HID * HID, HID * HID);
  cvt(d_in[13], wp, OUT_DIM * HID);

  lstm_scan_kernel<<<BATCH / 16, 256, 0, stream>>>(
      x, wx, wh, wp,
      (const float*)d_in[2],  (const float*)d_in[5],
      (const float*)d_in[8],  (const float*)d_in[11],
      (const float*)d_in[14], (float*)d_out);
}
